// Quantisation_16183436772095
// MI455X (gfx1250) — matrix operand order repaired and checked by reading, not compiled, not run on hardware
//
#include <hip/hip_runtime.h>
#include <hip/hip_bf16.h>

typedef __attribute__((ext_vector_type(16))) _Float16 v16h;
typedef __attribute__((ext_vector_type(8)))  _Float16 v8h;
typedef __attribute__((ext_vector_type(8)))  float    v8f;

#define NUM_EMB 2048
#define EMB_DIM 256
#define BATCH   64
#define HW      1024          // 32*32
#define NVEC    (BATCH * HW)  // 65536
#define BPAD    264           // 256 halves + 8 pad -> per-code bank offset of 4 dwords
#define KTILE   32            // codes per iteration (two 16-code WMMA sub-tiles)
#define NITER   (NUM_EMB / KTILE)  // 64

// ---------------------------------------------------------------------------
// Kernel 1: emb f32 -> f16, and e_sq[k] = sum_c emb[k][c]^2
// ---------------------------------------------------------------------------
__global__ void prep_emb(const float* __restrict__ emb,
                         _Float16* __restrict__ emb_h,
                         float* __restrict__ e_sq) {
    const int k = blockIdx.x;
    const int c = threadIdx.x;
    float v = emb[(size_t)k * EMB_DIM + c];
    emb_h[(size_t)k * EMB_DIM + c] = (_Float16)v;
    float s = v * v;
#pragma unroll
    for (int off = 16; off > 0; off >>= 1)
        s += __shfl_down(s, off, 32);
    __shared__ float partial[8];
    if ((threadIdx.x & 31) == 0) partial[threadIdx.x >> 5] = s;
    __syncthreads();
    if (threadIdx.x == 0) {
        float t = 0.f;
#pragma unroll
        for (int i = 0; i < 8; ++i) t += partial[i];
        e_sq[k] = t;
    }
}

// ---------------------------------------------------------------------------
// Kernel 2: NCHW f32 -> channels-last [N, C] f16 via LDS tile transpose
// ---------------------------------------------------------------------------
__global__ void transpose_x(const float* __restrict__ x,
                            _Float16* __restrict__ x_t) {
    __shared__ float tile[32][33];
    const int b  = blockIdx.z;
    const int ct = blockIdx.y;
    const int ht = blockIdx.x;
    const int tx = threadIdx.x;  // 0..31
    const int ty = threadIdx.y;  // 0..7

    const float* src = x + ((size_t)b * EMB_DIM + (size_t)ct * 32) * HW + (size_t)ht * 32;
#pragma unroll
    for (int i = 0; i < 4; ++i) {
        const int c = ty + 8 * i;
        tile[c][tx] = src[(size_t)c * HW + tx];   // coalesced over hw
    }
    __syncthreads();

    _Float16* dst = x_t + ((size_t)b * HW + (size_t)ht * 32) * EMB_DIM + (size_t)ct * 32;
#pragma unroll
    for (int i = 0; i < 4; ++i) {
        const int r = ty + 8 * i;
        dst[(size_t)r * EMB_DIM + tx] = (_Float16)tile[tx][r];  // coalesced over c
    }
}

// ---------------------------------------------------------------------------
// Kernel 3: WMMA argmin.  512 blocks x 256 threads (8 waves).
// Each wave owns 16 rows; 64 iterations of 32 codes (two independent
// accumulator chains, double-buffered LDS staging, one barrier per iter).
// Score = e_sq[k] - 2 * (x . e_k)   (||x||^2 dropped: constant per row)
// ---------------------------------------------------------------------------
__global__ __launch_bounds__(256) void vq_argmin(const _Float16* __restrict__ xt,
                                                 const _Float16* __restrict__ eh,
                                                 const float* __restrict__ esq,
                                                 int* __restrict__ idx_out) {
    __shared__ _Float16 bsm[2][KTILE * BPAD];  // 2 x 32 codes x 256 ch (+pad) = 33792 B
    __shared__ float    esm[NUM_EMB];          // 8 KB

    const int tid  = threadIdx.x;
    const int lane = tid & 31;
    const int wave = tid >> 5;
    const int col  = lane & 15;   // code-in-subtile / row-in-A
    const int hsel = lane >> 4;   // half-wave select
    const int rowBase = blockIdx.x * 128 + wave * 16;

    // --- A fragment: 16 rows x 256 ch f16 held in registers for all K tiles.
    // ISA 16-bit A layout: low lanes K={0..7,16..23}, high lanes K={8..15,24..31}.
    v16h a[8];
    {
        const _Float16* arow = xt + (size_t)(rowBase + col) * EMB_DIM;
#pragma unroll
        for (int cc = 0; cc < 8; ++cc) {
            const int o = cc * 32 + hsel * 8;
            v8h lo = *(const v8h*)(arow + o);
            v8h hi = *(const v8h*)(arow + o + 16);
            a[cc] = __builtin_shufflevector(lo, hi,
                     0, 1, 2, 3, 4, 5, 6, 7, 8, 9, 10, 11, 12, 13, 14, 15);
        }
    }

    float minv[8];
    int   mini[8];
#pragma unroll
    for (int g = 0; g < 8; ++g) { minv[g] = 3.4e38f; mini[g] = 0; }

    // cooperative staging map: 256 threads move 32 codes x 256 halves (4xb128 each)
    const int scode  = tid >> 3;        // 0..31
    const int schunk = (tid & 7) * 32;  // 0..224 step 32 (halves)

    // stage e_sq into LDS (2048 floats) + first emb tile (buffer 0)
#pragma unroll
    for (int i = 0; i < 8; ++i)
        esm[tid + 256 * i] = esq[tid + 256 * i];
    {
        const _Float16* g0 = eh + (size_t)scode * EMB_DIM + schunk;
        _Float16* s0 = &bsm[0][scode * BPAD + schunk];
#pragma unroll
        for (int i = 0; i < 4; ++i)
            *(v8h*)(s0 + 8 * i) = *(const v8h*)(g0 + 8 * i);
    }
    __syncthreads();

    for (int kt = 0; kt < NITER; ++kt) {
        const int buf = kt & 1;
        const int k0  = kt * KTILE;

        // stage next tile into the other buffer while this one computes
        if (kt + 1 < NITER) {
            const _Float16* g0 = eh + (size_t)(k0 + KTILE + scode) * EMB_DIM + schunk;
            _Float16* s0 = &bsm[buf ^ 1][scode * BPAD + schunk];
#pragma unroll
            for (int i = 0; i < 4; ++i)
                *(v8h*)(s0 + 8 * i) = *(const v8h*)(g0 + 8 * i);
        }

        // two independent WMMA accumulator chains (codes k0..k0+15, k0+16..k0+31)
        v8f acc0 = {};
        v8f acc1 = {};
#pragma unroll
        for (int cc = 0; cc < 8; ++cc) {
            // ISA 16-bit B layout: low lanes K={0..7,16..23}, high lanes K={8..15,24..31}
            const _Float16* bp0 = &bsm[buf][col * BPAD + cc * 32 + hsel * 8];
            const _Float16* bp1 = bp0 + 16 * BPAD;
            v8h l0 = *(const v8h*)(bp0);
            v8h h0 = *(const v8h*)(bp0 + 16);
            v8h l1 = *(const v8h*)(bp1);
            v8h h1 = *(const v8h*)(bp1 + 16);
            v16h b0 = __builtin_shufflevector(l0, h0,
                       0, 1, 2, 3, 4, 5, 6, 7, 8, 9, 10, 11, 12, 13, 14, 15);
            v16h b1 = __builtin_shufflevector(l1, h1,
                       0, 1, 2, 3, 4, 5, 6, 7, 8, 9, 10, 11, 12, 13, 14, 15);
            acc0 = __builtin_amdgcn_wmma_f32_16x16x32_f16(
                       false, a[cc], false, b0, (short)0, acc0, false, false);
            acc1 = __builtin_amdgcn_wmma_f32_16x16x32_f16(
                       false, a[cc], false, b1, (short)0, acc1, false, false);
        }

        const float es0 = esm[k0 + col];
        const float es1 = esm[k0 + 16 + col];
        const int   ka  = k0 + col;
        const int   kb  = k0 + 16 + col;
#pragma unroll
        for (int g = 0; g < 8; ++g) {
            const float s0 = __builtin_fmaf(-2.0f, acc0[g], es0);
            if (s0 < minv[g]) { minv[g] = s0; mini[g] = ka; }
            const float s1 = __builtin_fmaf(-2.0f, acc1[g], es1);
            if (s1 < minv[g]) { minv[g] = s1; mini[g] = kb; }
        }

        // one barrier per iteration: waits our ds traffic, then publishes
        // buf^1 stores and protects buf from next iteration's overwrite.
        __syncthreads();
    }

    // min/argmin butterfly across the 16 lanes sharing each row.
    // Tie-break toward lower index, matching jnp.argmin semantics.
#pragma unroll
    for (int g = 0; g < 8; ++g) {
        float v = minv[g];
        int   i = mini[g];
#pragma unroll
        for (int off = 1; off < 16; off <<= 1) {
            const float ov = __shfl_xor(v, off, 32);
            const int   oi = __shfl_xor(i, off, 32);
            if (ov < v || (ov == v && oi < i)) { v = ov; i = oi; }
        }
        if (col == 0) idx_out[rowBase + g + 8 * hsel] = i;
    }
}

// ---------------------------------------------------------------------------
// Kernel 4: out[b][c][hw] = emb[idx[b*HW+hw]][c]  (f32, NCHW)
// ---------------------------------------------------------------------------
__global__ void gather_out(const int* __restrict__ idx,
                           const float* __restrict__ emb,
                           float* __restrict__ out) {
    const int b  = blockIdx.x >> 2;
    const int hc = blockIdx.x & 3;
    const int hw = hc * 256 + threadIdx.x;
    const int id = idx[b * HW + hw];
    const float* er = emb + (size_t)id * EMB_DIM;
    float* o = out + (size_t)b * EMB_DIM * HW + hw;
#pragma unroll 4
    for (int c = 0; c < EMB_DIM; ++c)
        o[(size_t)c * HW] = er[c];   // coalesced writes; reads are L2-resident gathers
}

extern "C" void kernel_launch(void* const* d_in, const int* in_sizes, int n_in,
                              void* d_out, int out_size, void* d_ws, size_t ws_size,
                              hipStream_t stream) {
    const float* x   = (const float*)d_in[0];   // [64, 256, 32, 32] f32
    const float* emb = (const float*)d_in[1];   // [2048, 256] f32
    float* out = (float*)d_out;                 // [64, 256, 32, 32] f32

    char* ws = (char*)d_ws;
    const size_t OFF_XT  = 0;                                      // 32 MB  f16 [N, C]
    const size_t OFF_EH  = OFF_XT + (size_t)NVEC * EMB_DIM * 2;    //  1 MB  f16 [K, C]
    const size_t OFF_ESQ = OFF_EH + (size_t)NUM_EMB * EMB_DIM * 2; //  8 KB  f32 [K]
    const size_t OFF_IDX = OFF_ESQ + (size_t)NUM_EMB * 4;          // 256 KB i32 [N]
    _Float16* x_t   = (_Float16*)(ws + OFF_XT);
    _Float16* emb_h = (_Float16*)(ws + OFF_EH);
    float*    e_sq  = (float*)(ws + OFF_ESQ);
    int*      idxb  = (int*)(ws + OFF_IDX);

    prep_emb<<<NUM_EMB, 256, 0, stream>>>(emb, emb_h, e_sq);
    transpose_x<<<dim3(32, 8, BATCH), dim3(32, 8), 0, stream>>>(x, x_t);
    vq_argmin<<<NVEC / 128, 256, 0, stream>>>(x_t, emb_h, e_sq, idxb);
    gather_out<<<BATCH * 4, 256, 0, stream>>>(idxb, emb, out);
}
